// TransformerBlock_40621800686052
// MI455X (gfx1250) — matrix operand order repaired and checked by reading, not compiled, not run on hardware
//
#include <hip/hip_runtime.h>
#include <hip/hip_bf16.h>
#include <math.h>

// ---------------------------------------------------------------------------
// MI455X (gfx1250) transformer block.
// Compute-bound (~275 GFLOP vs ~112 MB min traffic -> AI ~2400 FLOP/B): all
// GEMMs (incl. attention) run on v_wmma_f32_16x16x32_bf16 with f32 accum.
// wave32; 128x128 block tile, 32x64 wave tile (8 WMMA per 32-K step/wave).
// A-tile LDS fill uses GLOBAL_LOAD_ASYNC_TO_LDS_B128 (ASYNCcnt path) — the
// builtin is present on this toolchain; params are (as1 v4i*, as3 v4i*,
// imm offset, imm cpol) per the round-2 diagnostic.
// ---------------------------------------------------------------------------

typedef __attribute__((ext_vector_type(16))) __bf16 v16bf;
typedef __attribute__((ext_vector_type(8)))  __bf16 v8bf;
typedef __attribute__((ext_vector_type(8)))  float  v8f;

#define DIM   1024
#define HEADS 16
#define HD    64
#define SEQ   2048
#define DFF   4096
#define MROWS 8192      // B*S

#if defined(__has_builtin)
#  if __has_builtin(__builtin_amdgcn_global_load_async_to_lds_b128) && \
      __has_builtin(__builtin_amdgcn_s_wait_asynccnt)
#    define USE_ASYNC_LDS 1
#  endif
#endif
#ifndef USE_ASYNC_LDS
#  define USE_ASYNC_LDS 0
#endif

#if USE_ASYNC_LDS
typedef int v4i_vs __attribute__((vector_size(16)));
typedef __attribute__((address_space(1))) v4i_vs* gvec_p;   // global src
typedef __attribute__((address_space(3))) v4i_vs* lvec_p;   // LDS dst
#endif

__device__ __forceinline__ unsigned short f2bf(float f) {
  union { float f; unsigned u; } c; c.f = f;
  unsigned r = c.u + 0x7FFFu + ((c.u >> 16) & 1u);   // round-to-nearest-even
  return (unsigned short)(r >> 16);
}

__device__ __forceinline__ v16bf make_frag(uint4 lo, uint4 hi) {
  v8bf l = __builtin_bit_cast(v8bf, lo);
  v8bf h = __builtin_bit_cast(v8bf, hi);
  v16bf r;
#pragma unroll
  for (int j = 0; j < 8; ++j) { r[j] = l[j]; r[8 + j] = h[j]; }
  return r;
}

// ------------------------------- f32 -> bf16 -------------------------------
__global__ __launch_bounds__(256) void cvt_bf16(const float* __restrict__ in,
                                                unsigned short* __restrict__ out,
                                                int n) {
  int i = blockIdx.x * 256 + threadIdx.x;
  if (i < n) out[i] = f2bf(in[i]);
}

// ------------------------------- LayerNorm ---------------------------------
__global__ __launch_bounds__(256) void layernorm_bf16(
    const float* __restrict__ x, const float* __restrict__ sc,
    const float* __restrict__ bi, unsigned short* __restrict__ out) {
  __shared__ float red[2][256];
  const int row = blockIdx.x;
  const float* xr = x + (size_t)row * DIM;
  float v[4], s = 0.f, s2 = 0.f;
#pragma unroll
  for (int j = 0; j < 4; ++j) {
    v[j] = xr[threadIdx.x + 256 * j];
    s += v[j]; s2 += v[j] * v[j];
  }
  red[0][threadIdx.x] = s; red[1][threadIdx.x] = s2;
  __syncthreads();
  for (int off = 128; off > 0; off >>= 1) {
    if (threadIdx.x < off) {
      red[0][threadIdx.x] += red[0][threadIdx.x + off];
      red[1][threadIdx.x] += red[1][threadIdx.x + off];
    }
    __syncthreads();
  }
  float mu  = red[0][0] * (1.0f / DIM);
  float var = red[1][0] * (1.0f / DIM) - mu * mu;
  float rs  = rsqrtf(var + 1e-6f);
#pragma unroll
  for (int j = 0; j < 4; ++j) {
    int c = threadIdx.x + 256 * j;
    out[(size_t)row * DIM + c] = f2bf((v[j] - mu) * rs * sc[c] + bi[c]);
  }
}

// --------------------------- tiled bf16 WMMA GEMM --------------------------
// C[M,N] = A[M,K]*W[K,N] (+bias, epilogue by MODE)
// block tile 128x128, 8 waves (4x2), wave tile 32x64, K step 32 through LDS.
// MODE 0: QKV scatter -> q/k/v [B,H,S,64] bf16
// MODE 1: f32 out + residual add
// MODE 2: tanh-GELU -> bf16 out
#define BM 128
#define BN 128
#define BK 32
#define LPAD 48   // row stride (elements): 96B, 16B-aligned, breaks bank conflicts

template <int MODE>
__global__ __launch_bounds__(256) void gemm_bf16_wmma(
    const unsigned short* __restrict__ A, const unsigned short* __restrict__ W,
    const float* __restrict__ bias, const float* __restrict__ res,
    float* __restrict__ outF, unsigned short* __restrict__ outB0,
    unsigned short* __restrict__ outB1, unsigned short* __restrict__ outB2,
    int N, int K) {
  __shared__ unsigned short sA[BM][LPAD];   // row-major A tile (cols 0..31)
  __shared__ unsigned short sB[BN][LPAD];   // W tile TRANSPOSED: [n][k]

  const int tid  = threadIdx.x;
  const int wave = tid >> 5, lane = tid & 31;
  const int half = lane >> 4, l16 = lane & 15;
  const int wm = wave & 3, wn = wave >> 2;          // 4x2 wave grid
  const int m0 = blockIdx.y * BM, n0 = blockIdx.x * BN;

  v8f acc[2][4] = {};

  for (int k0 = 0; k0 < K; k0 += BK) {
    // ---- A tile: 512 x 16B, 2 per thread (coalesced) ----
#pragma unroll
    for (int p = 0; p < 2; ++p) {
      int idx = tid + p * 256;
      int r = idx >> 2, seg = idx & 3;
      const unsigned short* src = &A[(size_t)(m0 + r) * K + k0 + seg * 8];
#if USE_ASYNC_LDS
      __builtin_amdgcn_global_load_async_to_lds_b128(
          (gvec_p)src, (lvec_p)&sA[r][seg * 8], 0, 0);
#else
      *(uint4*)&sA[r][seg * 8] = *(const uint4*)src;
#endif
    }
    // ---- W tile: coalesced 16B loads, transposed scatter to LDS ----
#pragma unroll
    for (int p = 0; p < 2; ++p) {
      int idx = tid + p * 256;
      int r = idx >> 4, g = idx & 15;               // r: k row, g: n group
      uint4 v = *(const uint4*)&W[(size_t)(k0 + r) * N + n0 + g * 8];
      const unsigned short* e = (const unsigned short*)&v;
#pragma unroll
      for (int j = 0; j < 8; ++j) sB[g * 8 + j][r] = e[j];
    }
    if (k0 + BK < K) {  // pull next tiles toward L2 (global_prefetch_b8)
      __builtin_prefetch(&A[(size_t)(m0 + (tid >> 2)) * K + k0 + BK], 0, 1);
      __builtin_prefetch(&W[(size_t)(k0 + BK + (tid >> 4)) * N + n0], 0, 1);
    }
#if USE_ASYNC_LDS
    __builtin_amdgcn_s_wait_asynccnt(0);
#endif
    __syncthreads();

    // A fragments: lane l16 = row, element i -> k = (i%8) + half*8 + 16*(i/8)
    v16bf af[2], bfb[4];
#pragma unroll
    for (int ms = 0; ms < 2; ++ms) {
      int r = wm * 32 + ms * 16 + l16;
      uint4 lo = *(const uint4*)&sA[r][half * 8];
      uint4 hi = *(const uint4*)&sA[r][16 + half * 8];
      af[ms] = make_frag(lo, hi);
    }
    // B fragments: lane l16 = col n, element i -> k = (i%8) + half*8 + 16*(i/8)
#pragma unroll
    for (int ns = 0; ns < 4; ++ns) {
      int n = wn * 64 + ns * 16 + l16;
      uint4 lo = *(const uint4*)&sB[n][half * 8];
      uint4 hi = *(const uint4*)&sB[n][16 + half * 8];
      bfb[ns] = make_frag(lo, hi);
    }
#pragma unroll
    for (int ms = 0; ms < 2; ++ms)
#pragma unroll
      for (int ns = 0; ns < 4; ++ns)
        acc[ms][ns] = __builtin_amdgcn_wmma_f32_16x16x32_bf16(
            false, af[ms], false, bfb[ns], (short)0, acc[ms][ns], false, false);
    __syncthreads();
  }

  // epilogue: C layout -> row = base + half*8 + r, col = base + l16
#pragma unroll
  for (int ms = 0; ms < 2; ++ms)
#pragma unroll
    for (int ns = 0; ns < 4; ++ns)
#pragma unroll
      for (int r = 0; r < 8; ++r) {
        int gm = m0 + wm * 32 + ms * 16 + half * 8 + r;
        int gn = n0 + wn * 64 + ns * 16 + l16;
        float v = acc[ms][ns][r] + bias[gn];
        if (MODE == 0) {                       // QKV scatter to [B,H,S,64]
          int which = gn >> 10, rem = gn & 1023;
          int head = rem >> 6, hd = rem & 63;
          int b = gm >> 11, s = gm & 2047;
          size_t dst = ((((size_t)b * HEADS + head) * SEQ) + s) * HD + hd;
          unsigned short bv = f2bf(v);
          (which == 0 ? outB0 : which == 1 ? outB1 : outB2)[dst] = bv;
        } else if (MODE == 1) {                // f32 + residual
          size_t o = (size_t)gm * N + gn;
          outF[o] = v + res[o];
        } else {                               // tanh-GELU: v*sigmoid(2u)
          float u = 0.7978845608028654f * (v + 0.044715f * v * v * v);
          float g = v / (1.0f + __expf(-2.0f * u));
          outB0[(size_t)gm * N + gn] = f2bf(g);
        }
      }
}

// ------------------------------ flash attention ----------------------------
// grid (S/128, H, B); block 256 threads = 8 waves; each wave owns 16 q rows.
// Q fragments live in registers the whole kernel; K fragments stream straight
// from global (per-lane two 16B loads, L2-resident given 192MB L2);
// V staged transposed in LDS; online softmax via ds_bpermute reductions.
__global__ __launch_bounds__(256) void flash_attn_wmma(
    const unsigned short* __restrict__ Q, const unsigned short* __restrict__ Kb,
    const unsigned short* __restrict__ Vb, unsigned short* __restrict__ O) {
  __shared__ unsigned short sV[HD][LPAD];        // [hd][key], cols 0..31
  __shared__ unsigned short sP[8][16][LPAD];     // per-wave P transpose scratch

  const int tid  = threadIdx.x;
  const int wave = tid >> 5, lane = tid & 31;
  const int half = lane >> 4, l16 = lane & 15;
  const size_t base = ((size_t)blockIdx.z * HEADS + blockIdx.y) * SEQ * HD;
  const int q0 = blockIdx.x * 128 + wave * 16;

  // Q fragments over hd (two K=32 chunks), row = q0 + l16
  v16bf qf[2];
  {
    const unsigned short* qr = Q + base + (size_t)(q0 + l16) * HD;
#pragma unroll
    for (int c = 0; c < 2; ++c) {
      uint4 lo = *(const uint4*)&qr[c * 32 + half * 8];
      uint4 hi = *(const uint4*)&qr[c * 32 + 16 + half * 8];
      qf[c] = make_frag(lo, hi);
    }
  }

  v8f o_acc[4] = {};
  float m_i[8], l_i[8];
#pragma unroll
  for (int r = 0; r < 8; ++r) { m_i[r] = -1e30f; l_i[r] = 0.f; }
  const float scale = 0.125f;   // 64^-0.5

  for (int kt = 0; kt < SEQ; kt += 32) {
    // cooperative V tile -> sV transposed ([hd][key])
    {
      int r = tid >> 3, g = tid & 7;
      uint4 v = *(const uint4*)&Vb[base + (size_t)(kt + r) * HD + g * 8];
      const unsigned short* e = (const unsigned short*)&v;
#pragma unroll
      for (int j = 0; j < 8; ++j) sV[g * 8 + j][r] = e[j];
    }
    __syncthreads();

    // scores S = Q K^T : 16 q rows x 32 keys, K-dim = hd = 64 (2 WMMA chunks)
    v8f sc_t[2] = {};
#pragma unroll
    for (int ns = 0; ns < 2; ++ns) {
      const unsigned short* kr = Kb + base + (size_t)(kt + ns * 16 + l16) * HD;
#pragma unroll
      for (int c = 0; c < 2; ++c) {
        uint4 lo = *(const uint4*)&kr[c * 32 + half * 8];
        uint4 hi = *(const uint4*)&kr[c * 32 + 16 + half * 8];
        v16bf kf = make_frag(lo, hi);
        sc_t[ns] = __builtin_amdgcn_wmma_f32_16x16x32_bf16(
            false, qf[c], false, kf, (short)0, sc_t[ns], false, false);
      }
    }

    // online softmax; row of element r lives on the 16 lanes of this half
    float p_t[2][8], alpha[8];
#pragma unroll
    for (int r = 0; r < 8; ++r) {
      float mx = fmaxf(sc_t[0][r], sc_t[1][r]) * scale;
#pragma unroll
      for (int off = 1; off < 16; off <<= 1)
        mx = fmaxf(mx, __shfl_xor(mx, off, 32));
      float m_new = fmaxf(m_i[r], mx);
      alpha[r] = __expf(m_i[r] - m_new);
      float p0 = __expf(sc_t[0][r] * scale - m_new);
      float p1 = __expf(sc_t[1][r] * scale - m_new);
      p_t[0][r] = p0; p_t[1][r] = p1;
      float ls = p0 + p1;
#pragma unroll
      for (int off = 1; off < 16; off <<= 1)
        ls += __shfl_xor(ls, off, 32);
      l_i[r] = l_i[r] * alpha[r] + ls;
      m_i[r] = m_new;
    }
#pragma unroll
    for (int t = 0; t < 4; ++t)
#pragma unroll
      for (int r = 0; r < 8; ++r) o_acc[t][r] *= alpha[r];

    // P: C layout -> A layout via per-wave LDS (LDS is in-order per wave)
#pragma unroll
    for (int ns = 0; ns < 2; ++ns)
#pragma unroll
      for (int r = 0; r < 8; ++r)
        sP[wave][half * 8 + r][ns * 16 + l16] = f2bf(p_t[ns][r]);
    v16bf pf;
    {
      uint4 lo = *(const uint4*)&sP[wave][l16][half * 8];
      uint4 hi = *(const uint4*)&sP[wave][l16][16 + half * 8];
      pf = make_frag(lo, hi);
    }

    // O += P * V  (4 hd tiles of 16)
#pragma unroll
    for (int t = 0; t < 4; ++t) {
      int n = t * 16 + l16;
      uint4 lo = *(const uint4*)&sV[n][half * 8];
      uint4 hi = *(const uint4*)&sV[n][16 + half * 8];
      v16bf vf = make_frag(lo, hi);
      o_acc[t] = __builtin_amdgcn_wmma_f32_16x16x32_bf16(
          false, pf, false, vf, (short)0, o_acc[t], false, false);
    }
    __syncthreads();   // before next V tile overwrite
  }

  // write attn-out as [B,S,D] (D index = h*64 + hd) for the proj GEMM
#pragma unroll
  for (int t = 0; t < 4; ++t)
#pragma unroll
    for (int r = 0; r < 8; ++r) {
      int s = q0 + half * 8 + r;
      int hd = t * 16 + l16;
      float v = o_acc[t][r] / l_i[r];
      O[((size_t)blockIdx.z * SEQ + s) * DIM + blockIdx.y * HD + hd] = f2bf(v);
    }
}

// ------------------------------- launcher ----------------------------------
extern "C" void kernel_launch(void* const* d_in, const int* in_sizes, int n_in,
                              void* d_out, int out_size, void* d_ws,
                              size_t ws_size, hipStream_t stream) {
  const float* x     = (const float*)d_in[0];
  const float* ln1s  = (const float*)d_in[1];
  const float* ln1b  = (const float*)d_in[2];
  const float* ln2s  = (const float*)d_in[3];
  const float* ln2b  = (const float*)d_in[4];
  const float* qkvw  = (const float*)d_in[5];
  const float* qkvb  = (const float*)d_in[6];
  const float* projw = (const float*)d_in[7];
  const float* projb = (const float*)d_in[8];
  const float* fc1w  = (const float*)d_in[9];
  const float* fc1b  = (const float*)d_in[10];
  const float* fc2w  = (const float*)d_in[11];
  const float* fc2b  = (const float*)d_in[12];

  char* ws = (char*)d_ws;
  size_t off = 0;
  auto alloc = [&](size_t bytes) {
    void* p = ws + off;
    off += (bytes + 255) & ~(size_t)255;
    return p;
  };
  unsigned short* wqkv = (unsigned short*)alloc((size_t)DIM * 3 * DIM * 2);
  unsigned short* wprj = (unsigned short*)alloc((size_t)DIM * DIM * 2);
  unsigned short* wfc1 = (unsigned short*)alloc((size_t)DIM * DFF * 2);
  unsigned short* wfc2 = (unsigned short*)alloc((size_t)DFF * DIM * 2);
  unsigned short* hbf  = (unsigned short*)alloc((size_t)MROWS * DIM * 2);
  unsigned short* qb   = (unsigned short*)alloc((size_t)MROWS * DIM * 2);
  unsigned short* kb   = (unsigned short*)alloc((size_t)MROWS * DIM * 2);
  unsigned short* vb   = (unsigned short*)alloc((size_t)MROWS * DIM * 2);
  unsigned short* aob  = (unsigned short*)alloc((size_t)MROWS * DIM * 2);
  float*          x1   = (float*)alloc((size_t)MROWS * DIM * 4);
  unsigned short* h2bf = (unsigned short*)alloc((size_t)MROWS * DIM * 2);
  unsigned short* a1bf = (unsigned short*)alloc((size_t)MROWS * DFF * 2);

  // 1) weights -> bf16
  cvt_bf16<<<(DIM * 3 * DIM + 255) / 256, 256, 0, stream>>>(qkvw, wqkv, DIM * 3 * DIM);
  cvt_bf16<<<(DIM * DIM + 255) / 256, 256, 0, stream>>>(projw, wprj, DIM * DIM);
  cvt_bf16<<<(DIM * DFF + 255) / 256, 256, 0, stream>>>(fc1w, wfc1, DIM * DFF);
  cvt_bf16<<<(DFF * DIM + 255) / 256, 256, 0, stream>>>(fc2w, wfc2, DFF * DIM);

  // 2) LN1 -> bf16
  layernorm_bf16<<<MROWS, 256, 0, stream>>>(x, ln1s, ln1b, hbf);

  // 3) QKV GEMM (scatter to [B,H,S,64])
  gemm_bf16_wmma<0><<<dim3(3 * DIM / BN, MROWS / BM), 256, 0, stream>>>(
      hbf, wqkv, qkvb, nullptr, nullptr, qb, kb, vb, 3 * DIM, DIM);

  // 4) flash attention
  flash_attn_wmma<<<dim3(SEQ / 128, HEADS, 4), 256, 0, stream>>>(qb, kb, vb, aob);

  // 5) proj GEMM + residual -> x1 (f32)
  gemm_bf16_wmma<1><<<dim3(DIM / BN, MROWS / BM), 256, 0, stream>>>(
      aob, wprj, projb, x, x1, nullptr, nullptr, nullptr, DIM, DIM);

  // 6) LN2 -> bf16
  layernorm_bf16<<<MROWS, 256, 0, stream>>>(x1, ln2s, ln2b, h2bf);

  // 7) FC1 + GELU -> bf16
  gemm_bf16_wmma<2><<<dim3(DFF / BN, MROWS / BM), 256, 0, stream>>>(
      h2bf, wfc1, fc1b, nullptr, nullptr, a1bf, nullptr, nullptr, DFF, DIM);

  // 8) FC2 + residual -> out (f32)
  gemm_bf16_wmma<1><<<dim3(DIM / BN, MROWS / BM), 256, 0, stream>>>(
      a1bf, wfc2, fc2b, x1, (float*)d_out, nullptr, nullptr, nullptr, DIM, DFF);
}
